// Reservoir_6193342841633
// MI455X (gfx1250) — compile-verified, matrix operand order checked by reading, not run on hardware
//
#include <hip/hip_runtime.h>

// ---------------- problem constants ----------------
#define UNITS   2048
#define IN_DIM  128
#define KTOT    (UNITS + IN_DIM)     // 2176 fused K: [W | Win]
#define T_STEPS 4096
#define NCHUNK  (KTOT / 32)          // 68 bf16 WMMA K-chunks
#define NTILE   (UNITS / 16)         // 128 output tiles of 16 units
#define TILES_PER_BLOCK 4
#define NBLOCK  (NTILE / TILES_PER_BLOCK)  // 32 persistent workgroups
#define THREADS 128                  // 4 waves -> 1 tile per wave
#define FRAG_BYTES 1024              // one 16x32 bf16 A-fragment (32B/lane)
#define W_BYTES ((size_t)NTILE * NCHUNK * FRAG_BYTES)  // 8,912,896

typedef __bf16        v16bf __attribute__((ext_vector_type(16)));
typedef float         v8f   __attribute__((ext_vector_type(8)));
typedef unsigned int  u32;
typedef u32           u32x4 __attribute__((ext_vector_type(4)));

__device__ __forceinline__ unsigned short f2bf(float f) {
  u32 u = __float_as_uint(f);
  u32 r = u + 0x7FFFu + ((u >> 16) & 1u);   // round-to-nearest-even
  return (unsigned short)(r >> 16);
}

// K index inside a 32-chunk for (dword v, lane-half h) per the CDNA5
// 16-bit A layout: V0..3 -> K={0..7}+8h, V4..7 -> K={16..23}+8h.
__device__ __forceinline__ int kmap(int v, int h) {
  return ((v < 4) ? 2 * v : 16 + 2 * (v - 4)) + 8 * h;
}

// ---------------- kernel 1: pack [W | Win] -> bf16 WMMA A-fragments ----------------
__global__ void pack_kernel(const float* __restrict__ W,
                            const float* __restrict__ Win,
                            const float* __restrict__ s0,
                            unsigned short* __restrict__ wp,       // packed fragments
                            unsigned short* __restrict__ stateBuf, // [2][UNITS] bf16
                            unsigned int*   __restrict__ counter) {
  size_t tid    = (size_t)blockIdx.x * blockDim.x + threadIdx.x;
  size_t stride = (size_t)gridDim.x * blockDim.x;
  size_t total  = (size_t)NTILE * NCHUNK * 512;   // bf16 elements
  for (size_t e = tid; e < total; e += stride) {
    int frag = (int)(e >> 9);          // 512 bf16 per fragment
    int r    = (int)(e & 511);
    int lane = r >> 4, idx = r & 15;
    int v = idx >> 1, off = idx & 1;
    int h = lane >> 4, m = lane & 15;
    int tile = frag / NCHUNK, c = frag % NCHUNK;
    int k   = c * 32 + kmap(v, h) + off;
    int row = tile * 16 + m;
    float val = (k < UNITS) ? W[(size_t)row * UNITS + k]
                            : Win[(size_t)row * IN_DIM + (k - UNITS)];
    wp[e] = f2bf(val);
  }
  if (tid < UNITS) stateBuf[tid] = f2bf(s0[tid]);  // step-0 state (buffer 0)
  if (tid == 0)    *counter = 0u;                  // reset grid barrier
}

// ---------------- kernel 2: persistent recurrence ----------------
// 32 blocks x 128 threads; each block owns 4 output tiles and caches its
// 4*68 KB slice of packed [W|Win] in LDS; 68 WMMAs per wave per step.
__global__ void __launch_bounds__(THREADS, 1)
reservoir_kernel(const float* __restrict__ useq,
                 const float* __restrict__ s0,
                 const unsigned short* __restrict__ wp,
                 unsigned short* __restrict__ stateBuf,  // [2][UNITS]
                 unsigned int*   __restrict__ counter,
                 float* __restrict__ out) {
  extern __shared__ char smem[];
  // [0, 278528)        : W fragments for this block's 4 tiles
  // [278528, 282880)   : sext = bf16 [state(2048) | u_t(128)]
  // [282880, 283136)   : ytile exchange, 4 waves x 16 floats
  unsigned short* wlds  = (unsigned short*)smem;
  unsigned short* sext  = (unsigned short*)(smem + TILES_PER_BLOCK * NCHUNK * FRAG_BYTES);
  float*          ytile = (float*)(smem + TILES_PER_BLOCK * NCHUNK * FRAG_BYTES + KTOT * 2);

  const int tix  = threadIdx.x;
  const int lane = tix & 31;
  const int wv   = tix >> 5;
  const int half = lane >> 4;
  const int tile = blockIdx.x * TILES_PER_BLOCK + wv;
  const int j    = tile * 16 + (lane & 15);   // unit updated by lanes 0..15

  // Stage this block's weight slice into LDS once (contiguous b128 copy).
  {
    const int nq = TILES_PER_BLOCK * NCHUNK * FRAG_BYTES / 16;  // 17408
    const u32x4* src = (const u32x4*)wp + (size_t)blockIdx.x * nq;
    u32x4* dst = (u32x4*)wlds;
    for (int i = tix; i < nq; i += THREADS) dst[i] = src[i];
  }

  float sprev = s0[j];   // fp32 running state kept in registers (lanes 0..15)
  const float LR = 0.1f, OMLR = 0.9f;

  for (int t = 0; t < T_STEPS; ++t) {
    // Refill sext = [bf16 state from buffer t&1 | bf16(u_t)]
    {
      const u32x4* s4 = (const u32x4*)(stateBuf + (size_t)(t & 1) * UNITS);
      u32x4* d4 = (u32x4*)sext;
      for (int i = tix; i < UNITS * 2 / 16; i += THREADS) d4[i] = s4[i];
      if (tix < IN_DIM) sext[UNITS + tix] = f2bf(useq[(size_t)t * IN_DIM + tix]);
    }
    __syncthreads();

    // Fused matvec: pre = [s | u] . [W | Win]^T rows of this tile.
    v8f acc = {};
    const char* abase = (const char*)wlds + (size_t)wv * NCHUNK * FRAG_BYTES + lane * 32;
    for (int c = 0; c < NCHUNK; ++c) {
      union { u32x4 q[2]; v16bf v; } A, B;
      const u32x4* aq = (const u32x4*)(abase + c * FRAG_BYTES);
      A.q[0] = aq[0];
      A.q[1] = aq[1];
      // Broadcast-B: every column n holds the same K-chunk of sext.
      B.q[0] = *(const u32x4*)((const char*)sext + (c * 32 + 8 * half) * 2);
      B.q[1] = *(const u32x4*)((const char*)sext + (c * 32 + 16 + 8 * half) * 2);
      acc = __builtin_amdgcn_wmma_f32_16x16x32_bf16(false, A.v, false, B.v,
                                                    (short)0, acc, false, false);
    }

    // D layout: lanes 0..15 hold y[0..7] in acc[0..7]; lanes 16..31 hold y[8..15].
    // Exchange through LDS so 16 lanes each own one unit.
    union { v8f f; u32x4 q[2]; } au; au.f = acc;
    u32x4* ys = (u32x4*)(ytile + wv * 16);
    if (lane == 0)  { ys[0] = au.q[0]; ys[1] = au.q[1]; }
    if (lane == 16) { ys[2] = au.q[0]; ys[3] = au.q[1]; }
    // same-wave DS ops are in order; compiler inserts s_wait_dscnt before use
    if (lane < 16) {
      float y    = ytile[wv * 16 + lane];
      float news = OMLR * sprev + LR * tanhf(y);
      sprev = news;
      out[(size_t)t * UNITS + j] = news;                       // history[t]
      stateBuf[(size_t)((t + 1) & 1) * UNITS + j] = f2bf(news);
      if (t == T_STEPS - 1) out[(size_t)T_STEPS * UNITS + j] = news;  // final_state
    }

    // Device-wide step barrier (two-phase monotonic counter).
    __threadfence();
    __syncthreads();
    if (tix == 0) {
      __hip_atomic_fetch_add(counter, 1u, __ATOMIC_ACQ_REL, __HIP_MEMORY_SCOPE_AGENT);
      const unsigned int target = (unsigned int)(t + 1) * NBLOCK;
      while (__hip_atomic_load(counter, __ATOMIC_ACQUIRE, __HIP_MEMORY_SCOPE_AGENT) < target) {
        __builtin_amdgcn_s_sleep(8);
      }
    }
    __syncthreads();
    __threadfence();
  }
}

// ---------------- launcher ----------------
extern "C" void kernel_launch(void* const* d_in, const int* in_sizes, int n_in,
                              void* d_out, int out_size, void* d_ws, size_t ws_size,
                              hipStream_t stream) {
  const float* useq = (const float*)d_in[0];   // [4096,128]
  const float* s0   = (const float*)d_in[1];   // [1,2048]
  const float* W    = (const float*)d_in[2];   // [2048,2048]
  const float* Win  = (const float*)d_in[3];   // [2048,128]
  float* out = (float*)d_out;                  // history [4096,2048] ++ final [2048]

  char* ws = (char*)d_ws;
  unsigned short* wp       = (unsigned short*)ws;                       // 8.9 MB packed
  unsigned short* stateBuf = (unsigned short*)(ws + W_BYTES);           // 8 KB double buffer
  unsigned int*   counter  = (unsigned int*)(ws + W_BYTES + 2 * UNITS * 2);

  pack_kernel<<<2048, 256, 0, stream>>>(W, Win, s0, wp, stateBuf, counter);

  const size_t smem = (size_t)TILES_PER_BLOCK * NCHUNK * FRAG_BYTES + KTOT * 2
                    + TILES_PER_BLOCK * 16 * sizeof(float);             // 283136 B
  (void)hipFuncSetAttribute(reinterpret_cast<const void*>(reservoir_kernel),
                            hipFuncAttributeMaxDynamicSharedMemorySize, (int)smem);
  reservoir_kernel<<<NBLOCK, THREADS, smem, stream>>>(useq, s0, wp, stateBuf, counter, out);
}
